// ComplexAttentionLayer_27839978012965
// MI455X (gfx1250) — matrix operand order repaired and checked by reading, not compiled, not run on hardware
//
#include <hip/hip_runtime.h>
#include <hip/hip_bf16.h>

typedef __attribute__((ext_vector_type(16))) _Float16 v16h;
typedef __attribute__((ext_vector_type(8)))  _Float16 v8h;
typedef __attribute__((ext_vector_type(4)))  _Float16 v4h;
typedef __attribute__((ext_vector_type(8)))  float    v8f;

#define BB 4
#define LL 1024
#define DD 512
#define HH 8
#define EE 64
#define MM (BB * LL)        // 4096 rows

// ---------------------------------------------------------------------------
// fp32 -> f16 conversion, 4 elements per thread
// ---------------------------------------------------------------------------
__global__ void cvt_f32_to_f16_kernel(const float* __restrict__ src,
                                      _Float16* __restrict__ dst, int n4) {
    int i = blockIdx.x * blockDim.x + threadIdx.x;
    if (i < n4) {
        float4 f = ((const float4*)src)[i];
        v4h h;
        h[0] = (_Float16)f.x;
        h[1] = (_Float16)f.y;
        h[2] = (_Float16)f.z;
        h[3] = (_Float16)f.w;
        ((v4h*)dst)[i] = h;
    }
}

// ---------------------------------------------------------------------------
// CDNA5 async memory -> LDS copy (ASYNCcnt path, CDNA5 ISA).
// Generic LDS pointers carry the LDS byte offset in their low 32 bits
// (ISA §10.2), so integer casts suffice.
// ---------------------------------------------------------------------------
__device__ __forceinline__ void async_ld_b32(void* lds_dst, const void* gsrc) {
    unsigned int       loff  = (unsigned int)(unsigned long long)lds_dst;
    unsigned long long gaddr = (unsigned long long)gsrc;
    asm volatile("global_load_async_to_lds_b32 %0, %1, off"
                 :: "v"(loff), "v"(gaddr)
                 : "memory");
}
__device__ __forceinline__ void wait_asynccnt0() {
    asm volatile("s_wait_asynccnt 0x0" ::: "memory");
}

// ---------------------------------------------------------------------------
// Complex GEMM:  y_r + i*y_i = (x_r + i*x_i) @ (w_r + i*w_i)^T + (b_r + i*b_i)
//   y_r = (x_r@w_r^T) - (x_i@w_i^T) + (b_r - b_i)   [two accumulators, fold in epilogue]
//   y_i = (x_i@w_r^T) + (x_r@w_i^T) + (b_r + b_i)
// M = gridDim.x*64 rows, N = 512, K = 512.  f16 WMMA, fp32 accumulate.
// Per-wave 32x32 tile; 8 waves -> 64x128 per workgroup.
// ---------------------------------------------------------------------------
__device__ __forceinline__ v16h load_a_frag(const _Float16* p, int hsel) {
    // A frag in contiguous k order (same as B): lanes 0-15 hold K 0-15, lanes 16-31 hold K 16-31.
    // hsel = (lane>>4)*16.
    v8h lo = *(const v8h*)(p + hsel);
    v8h hi = *(const v8h*)(p + 8 + hsel);
    return __builtin_shufflevector(lo, hi, 0, 1, 2, 3, 4, 5, 6, 7,
                                           8, 9, 10, 11, 12, 13, 14, 15);
}

__device__ __forceinline__ v8f wmma_f16(v16h a, v16h b, v8f c) {
    return __builtin_amdgcn_wmma_f32_16x16x32_f16(
        /*neg_a=*/false, a, /*neg_b=*/false, b,
        /*c_mod=*/(short)0, c, /*reuse_a=*/false, /*reuse_b=*/false);
}

__global__ void __launch_bounds__(256)
complex_gemm_f16_wmma(const _Float16* __restrict__ xr, const _Float16* __restrict__ xi,
                      const _Float16* __restrict__ wr, const _Float16* __restrict__ wi,
                      const float* __restrict__ br, const float* __restrict__ bi,
                      float* __restrict__ yr, float* __restrict__ yi) {
    const int tid  = threadIdx.x;
    const int lane = tid & 31;
    const int wave = tid >> 5;
    const int r    = lane & 15;
    const int hsel = (lane >> 4) << 3;          // 0 or 8 (elements)
    const int m_base = blockIdx.x * 64 + (wave & 1) * 32;
    const int n_base = blockIdx.y * 128 + (wave >> 1) * 32;

    const v8f vzero = {0.f, 0.f, 0.f, 0.f, 0.f, 0.f, 0.f, 0.f};
    v8f accP[2][2], accM[2][2], accI[2][2];
#pragma unroll
    for (int mt = 0; mt < 2; ++mt)
#pragma unroll
        for (int nt = 0; nt < 2; ++nt) {
            accP[mt][nt] = vzero; accM[mt][nt] = vzero; accI[mt][nt] = vzero;
        }

    const _Float16* arow[2];
    const _Float16* airow[2];
    size_t coloff[2];
#pragma unroll
    for (int mt = 0; mt < 2; ++mt) {
        size_t ro = (size_t)(m_base + mt * 16 + r) * DD;
        arow[mt]  = xr + ro;
        airow[mt] = xi + ro;
    }
#pragma unroll
    for (int nt = 0; nt < 2; ++nt)
        coloff[nt] = (size_t)(n_base + nt * 16 + r) * DD;

    const int bksel = (lane >> 4) << 4;         // 0 or 16 (elements) for B frags

    for (int k0 = 0; k0 < DD; k0 += 32) {
        // next-iteration prefetch (global_prefetch_b8)
        if (k0 + 32 < DD) {
            __builtin_prefetch(arow[0] + k0 + 32, 0, 0);
            __builtin_prefetch(airow[0] + k0 + 32, 0, 0);
        }
        v16h a_r[2], a_i[2];
#pragma unroll
        for (int mt = 0; mt < 2; ++mt) {
            a_r[mt] = load_a_frag(arow[mt] + k0, bksel);
            a_i[mt] = load_a_frag(airow[mt] + k0, bksel);
        }
        v16h b_r[2], b_i[2];
#pragma unroll
        for (int nt = 0; nt < 2; ++nt) {
            b_r[nt] = *(const v16h*)(wr + coloff[nt] + k0 + bksel);
            b_i[nt] = *(const v16h*)(wi + coloff[nt] + k0 + bksel);
        }
#pragma unroll
        for (int nt = 0; nt < 2; ++nt) {
#pragma unroll
            for (int mt = 0; mt < 2; ++mt) {
                accP[mt][nt] = wmma_f16(a_r[mt], b_r[nt], accP[mt][nt]);
                accM[mt][nt] = wmma_f16(a_i[mt], b_i[nt], accM[mt][nt]);
                accI[mt][nt] = wmma_f16(a_i[mt], b_r[nt], accI[mt][nt]);
                accI[mt][nt] = wmma_f16(a_r[mt], b_i[nt], accI[mt][nt]);
            }
        }
    }

    // epilogue: C/D 16x16 f32 layout -> lane r = col, vgpr j = row (+8 for hi half)
#pragma unroll
    for (int mt = 0; mt < 2; ++mt) {
#pragma unroll
        for (int nt = 0; nt < 2; ++nt) {
            const int col = n_base + nt * 16 + r;
            const float bR = br[col], bI = bi[col];
            const float addR = bR - bI, addI = bR + bI;
#pragma unroll
            for (int j = 0; j < 8; ++j) {
                const int row = m_base + mt * 16 + hsel + j;
                yr[(size_t)row * DD + col] = (accP[mt][nt][j] - accM[mt][nt][j]) + addR;
                yi[(size_t)row * DD + col] = accI[mt][nt][j] + addI;
            }
        }
    }
}

// ---------------------------------------------------------------------------
// Per-token complex "attention": for each token n and head h the score matrix
// is the rank-1 outer product over the E=64 axis:
//   s_r[l,s] = qr[l]kr[s] - qi[l]ki[s],  s_i[l,s] = qr[l]ki[s] - qi[l]kr[s]
//   attn = softmax_s( sqrt(s_r^2 + s_i^2) );  O[l] = sum_s attn[l,s] * v[s]
// One block per token, one wave32 per head; online softmax, 2 rows per lane.
// k/v head-vectors are staged into LDS with GLOBAL_LOAD_ASYNC_TO_LDS_B32.
// Writes f16 so the output projection consumes it directly.
// ---------------------------------------------------------------------------
__global__ void __launch_bounds__(256)
complex_attention_kernel(const float* __restrict__ Qr, const float* __restrict__ Qi,
                         const float* __restrict__ Kr, const float* __restrict__ Ki,
                         const float* __restrict__ Vr, const float* __restrict__ Vi,
                         _Float16* __restrict__ Or, _Float16* __restrict__ Oi) {
    __shared__ float sKr[HH][EE], sKi[HH][EE], sVr[HH][EE], sVi[HH][EE];
    const size_t base = (size_t)blockIdx.x * DD;
    const int tid = threadIdx.x;

    // async memory -> LDS staging (ASYNCcnt)
    for (int d = tid; d < DD; d += 256) {
        const int h = d & (HH - 1);
        const int e = d >> 3;
        async_ld_b32(&sKr[h][e], Kr + base + d);
        async_ld_b32(&sKi[h][e], Ki + base + d);
        async_ld_b32(&sVr[h][e], Vr + base + d);
        async_ld_b32(&sVi[h][e], Vi + base + d);
    }
    wait_asynccnt0();
    __syncthreads();

    const int h    = tid >> 5;
    const int lane = tid & 31;
    const int l0 = lane, l1 = lane + 32;

    const float qr0 = Qr[base + l0 * HH + h], qi0 = Qi[base + l0 * HH + h];
    const float qr1 = Qr[base + l1 * HH + h], qi1 = Qi[base + l1 * HH + h];

    float m0 = -INFINITY, S0 = 0.f, Ar0 = 0.f, Ai0 = 0.f;
    float m1 = -INFINITY, S1 = 0.f, Ar1 = 0.f, Ai1 = 0.f;

#pragma unroll 8
    for (int s = 0; s < EE; ++s) {
        const float krs = sKr[h][s], kis = sKi[h][s];
        const float vrs = sVr[h][s], vis = sVi[h][s];
        {
            const float sr = qr0 * krs - qi0 * kis;
            const float si = qr0 * kis - qi0 * krs;
            const float x  = sqrtf(sr * sr + si * si);
            const float mn = fmaxf(m0, x);
            const float sc = __expf(m0 - mn);
            const float p  = __expf(x - mn);
            S0  = S0  * sc + p;
            Ar0 = Ar0 * sc + p * vrs;
            Ai0 = Ai0 * sc + p * vis;
            m0 = mn;
        }
        {
            const float sr = qr1 * krs - qi1 * kis;
            const float si = qr1 * kis - qi1 * krs;
            const float x  = sqrtf(sr * sr + si * si);
            const float mn = fmaxf(m1, x);
            const float sc = __expf(m1 - mn);
            const float p  = __expf(x - mn);
            S1  = S1  * sc + p;
            Ar1 = Ar1 * sc + p * vrs;
            Ai1 = Ai1 * sc + p * vis;
            m1 = mn;
        }
    }

    Or[base + l0 * HH + h] = (_Float16)(Ar0 / S0);
    Oi[base + l0 * HH + h] = (_Float16)(Ai0 / S0);
    Or[base + l1 * HH + h] = (_Float16)(Ar1 / S1);
    Oi[base + l1 * HH + h] = (_Float16)(Ai1 / S1);
}

// ---------------------------------------------------------------------------
// Host-side orchestration
// ---------------------------------------------------------------------------
extern "C" void kernel_launch(void* const* d_in, const int* in_sizes, int n_in,
                              void* d_out, int out_size, void* d_ws, size_t ws_size,
                              hipStream_t stream) {
    (void)in_sizes; (void)n_in; (void)out_size; (void)ws_size;

    // inputs in setup_inputs() order
    const float* act[6];                  // q_r q_i k_r k_i v_r v_i
    for (int i = 0; i < 6; ++i) act[i] = (const float*)d_in[i];
    const float* w[8];                    // wq_r wq_i wk_r wk_i wv_r wv_i wo_r wo_i
    for (int i = 0; i < 8; ++i) w[i] = (const float*)d_in[6 + i];
    const float* bia[8];                  // bq_r bq_i bk_r bk_i bv_r bv_i bo_r bo_i
    for (int i = 0; i < 8; ++i) bia[i] = (const float*)d_in[14 + i];

    // workspace carve-up
    char* ws = (char*)d_ws;
    size_t off = 0;
    auto carve = [&](size_t bytes) -> void* {
        void* p = ws + off;
        off += (bytes + 255) & ~(size_t)255;
        return p;
    };
    const size_t MD = (size_t)MM * DD;    // 2,097,152 elements
    const size_t WD = (size_t)DD * DD;    //   262,144 elements

    _Float16* actH[6];
    for (int i = 0; i < 6; ++i) actH[i] = (_Float16*)carve(MD * sizeof(_Float16));
    // per linear (q,k,v,o): wr_h, wi_h
    _Float16* wH[4][2];
    for (int lin = 0; lin < 4; ++lin)
        for (int j = 0; j < 2; ++j) wH[lin][j] = (_Float16*)carve(WD * sizeof(_Float16));
    float* proj[6];                       // Qr Qi Kr Ki Vr Vi (fp32)
    for (int i = 0; i < 6; ++i) proj[i] = (float*)carve(MD * sizeof(float));
    _Float16* OrH = (_Float16*)carve(MD * sizeof(_Float16));
    _Float16* OiH = (_Float16*)carve(MD * sizeof(_Float16));

    // 1) activations -> f16
    {
        const int n4 = (int)(MD / 4);
        const int blocks = (n4 + 255) / 256;
        for (int i = 0; i < 6; ++i)
            cvt_f32_to_f16_kernel<<<blocks, 256, 0, stream>>>(act[i], actH[i], n4);
    }
    // 2) weights -> f16
    {
        const int n4 = (int)(WD / 4);
        const int blocks = (n4 + 255) / 256;
        for (int lin = 0; lin < 4; ++lin) {
            cvt_f32_to_f16_kernel<<<blocks, 256, 0, stream>>>(w[2 * lin + 0], wH[lin][0], n4);
            cvt_f32_to_f16_kernel<<<blocks, 256, 0, stream>>>(w[2 * lin + 1], wH[lin][1], n4);
        }
    }

    // 3) q/k/v complex projections (WMMA)
    dim3 gGrid(MM / 64, DD / 128);
    complex_gemm_f16_wmma<<<gGrid, 256, 0, stream>>>(
        actH[0], actH[1], wH[0][0], wH[0][1], bia[0], bia[1], proj[0], proj[1]);
    complex_gemm_f16_wmma<<<gGrid, 256, 0, stream>>>(
        actH[2], actH[3], wH[1][0], wH[1][1], bia[2], bia[3], proj[2], proj[3]);
    complex_gemm_f16_wmma<<<gGrid, 256, 0, stream>>>(
        actH[4], actH[5], wH[2][0], wH[2][1], bia[4], bia[5], proj[4], proj[5]);

    // 4) per-token complex attention (online softmax, async LDS staging)
    complex_attention_kernel<<<MM, 256, 0, stream>>>(
        proj[0], proj[1], proj[2], proj[3], proj[4], proj[5], OrH, OiH);

    // 5) output complex projection straight into d_out (out_r then out_i)
    float* out_r = (float*)d_out;
    float* out_i = out_r + MD;
    complex_gemm_f16_wmma<<<gGrid, 256, 0, stream>>>(
        OrH, OiH, wH[3][0], wH[3][1], bia[6], bia[7], out_r, out_i);
}
